// BaseConvFFF_15350213116006
// MI455X (gfx1250) — matrix operand order repaired and checked by reading, not compiled, not run on hardware
//
#include <hip/hip_runtime.h>

// ---------------------------------------------------------------------------
// CDNA5 WMMA (wave32): D(16x16,f32) = A(16x32,bf16) x B(32x16,bf16) + C
// A layout: lane = row M (%16); lo lanes elems 0..7 -> K=e, 8..15 -> K=e+8;
//           hi lanes +8.  B layout: lane = col N (%16); K order same as A.
// C layout: lane = N (%16); VGPR v -> M = v + (hi?8:0).
// ---------------------------------------------------------------------------
typedef __bf16 v16bf __attribute__((ext_vector_type(16)));
typedef float  v8f   __attribute__((ext_vector_type(8)));
typedef unsigned short ushort8 __attribute__((ext_vector_type(8)));

__device__ __forceinline__ unsigned short f2bf(float f) {
  unsigned u = __builtin_bit_cast(unsigned, f);
  u += 0x7FFFu + ((u >> 16) & 1u);          // round-to-nearest-even
  return (unsigned short)(u >> 16);
}
__device__ __forceinline__ __bf16 bfbits(unsigned short u) {
  return __builtin_bit_cast(__bf16, u);
}
__device__ __forceinline__ void fill_afrag(v16bf& a, ushort8 p0, ushort8 p1) {
#pragma unroll
  for (int e = 0; e < 8; ++e) { a[e] = bfbits(p0[e]); a[e + 8] = bfbits(p1[e]); }
}

// ---------------- problem constants ----------------
#define NB      32
#define INC     3
#define OUTC    64
#define HH      64
#define WW      64
#define H2      32
#define W2      32
#define NLEAF   16
#define LEAFW   128
#define OUTW    100

// ---------------- workspace layout (bytes) ----------------
#define OFF_W1R   ((size_t)0)                         // 98304   u16 (K padded to 96)
#define OFF_W2R   ((size_t)196608)                    // 1638400 u16 (tap-major K)
#define OFF_H     ((size_t)3473408)                   // 33554432 u16
#define OFF_FEAT  ((size_t)70582272)                  // 32768 f32 (zeroed each call)
#define OFF_MIX   ((size_t)70713344)                  // 512 f32

// ===========================================================================
// Prep: weight reorder/convert to bf16 + zero feat (atomicMax identity).
//  w1r[row][k96]  : k<75 -> cw1[row*75+k], else 0          (row = l*64+co)
//  w2r[row][t][ci]: <- cw2[(row*64+ci)*25 + t]  (K order = tap-major)
// ===========================================================================
__global__ void prep_kernel(const float* __restrict__ cw1,
                            const float* __restrict__ cw2,
                            unsigned short* __restrict__ w1r,
                            unsigned short* __restrict__ w2r,
                            float* __restrict__ feat) {
  const int n_w1 = NLEAF * OUTC * 96;                 // 98304
  const int n_w2 = NLEAF * OUTC * OUTC * 25;          // 1638400
  const int n_feat = NLEAF * NB * OUTC;               // 32768
  const int total = n_w1 + n_w2 + n_feat;
  for (int i = blockIdx.x * blockDim.x + threadIdx.x; i < total;
       i += gridDim.x * blockDim.x) {
    int j = i;
    if (j < n_w1) {
      int k = j % 96, row = j / 96;
      w1r[j] = (k < 75) ? f2bf(cw1[row * 75 + k]) : (unsigned short)0;
      continue;
    }
    j -= n_w1;
    if (j < n_w2) {
      int ci = j & 63, r = j >> 6;
      int t = r % 25, row = r / 25;
      w2r[j] = f2bf(cw2[(row * 64 + ci) * 25 + t]);
      continue;
    }
    j -= n_w2;
    feat[j] = 0.0f;
  }
}

// ===========================================================================
// Routing: 4 depth filters, conv over 64x64, per-batch max -> gates -> mix.
// ===========================================================================
__global__ __launch_bounds__(256)
void route_kernel(const float* __restrict__ x, const float* __restrict__ nw,
                  const float* __restrict__ nb, float* __restrict__ mix) {
  const int b = blockIdx.x, tid = threadIdx.x;
  __shared__ float wsm[4 * 75];
  __shared__ float red[256];
  __shared__ float score_sh[4];
  const int fidx[4] = {0, 2, 6, 14};
  for (int i = tid; i < 300; i += 256) wsm[i] = nw[fidx[i / 75] * 75 + (i % 75)];
  __syncthreads();
  float m[4] = {-1e30f, -1e30f, -1e30f, -1e30f};
  for (int p = tid; p < HH * WW; p += 256) {
    int y = p >> 6, xx = p & 63;
    float a[4] = {0.f, 0.f, 0.f, 0.f};
    for (int c = 0; c < INC; ++c)
      for (int ky = 0; ky < 5; ++ky) {
        int yy = y + ky - 2;
        if (yy < 0 || yy >= HH) continue;
        for (int kx = 0; kx < 5; ++kx) {
          int xc = xx + kx - 2;
          if (xc < 0 || xc >= WW) continue;
          float xv = x[((b * INC + c) * HH + yy) * WW + xc];
          int wi = (c * 5 + ky) * 5 + kx;
          a[0] += xv * wsm[wi];        a[1] += xv * wsm[75 + wi];
          a[2] += xv * wsm[150 + wi];  a[3] += xv * wsm[225 + wi];
        }
      }
    for (int d = 0; d < 4; ++d) m[d] = fmaxf(m[d], a[d]);
  }
  for (int d = 0; d < 4; ++d) {
    red[tid] = m[d]; __syncthreads();
    for (int s = 128; s > 0; s >>= 1) {
      if (tid < s) red[tid] = fmaxf(red[tid], red[tid + s]);
      __syncthreads();
    }
    if (tid == 0) score_sh[d] = red[0];
    __syncthreads();
  }
  if (tid < NLEAF) {
    int l = tid; float mm = 1.0f;
    for (int d = 0; d < 4; ++d) {
      int g = l >> (3 - d);
      int j = g >> 1, bit = g & 1;
      int plat = (1 << d) - 1;
      float be = 1.0f / (1.0f + __expf(-(score_sh[d] + nb[plat + j])));
      mm *= bit ? be : (1.0f - be);
    }
    mix[b * NLEAF + l] = mm;
  }
}

// ===========================================================================
// conv1 + maxpool2 + relu.  B tiles are staged in LDS *in fragment order*
// (index math done once during staging, from fp32 x directly); A rows padded
// to K=96 in prep.  Inner loop: 2 global b128 + 4 ds b128 + 2 wmma per step.
// Block = (leaf, batch, pooled row py, 16-wide x tile).
// ===========================================================================
__global__ __launch_bounds__(128)
void conv1_kernel(const float* __restrict__ x,
                  const unsigned short* __restrict__ w1r,
                  unsigned short* __restrict__ h) {
  const int l = blockIdx.z, b = blockIdx.y;
  const int py = blockIdx.x >> 2, xt = blockIdx.x & 3;
  const int x0 = xt * 16;
  const int tid = threadIdx.x;
  const int lane = tid & 31, wave = tid >> 5;
  // B tiles in fragment order: [yv(2)][kk(3)][lane(32)][e(16)] u16  (6 KB)
  __shared__ unsigned short bstage[2 * 3 * 32 * 16];
  for (int i = tid; i < 2 * 3 * 32 * 16; i += 128) {
    int e = i & 15, ln = (i >> 4) & 31, kv = i >> 9;   // kv = yv*3+kk
    int kk = kv % 3, yv = kv / 3;
    int n = ln & 15;
    int k = kk * 32 + ((e & 8) ? 16 : 0) + ((ln >= 16) ? 8 : 0) + (e & 7);       // dense-B K mapping
    unsigned short v = 0;
    if (k < 75) {
      int ci = k / 25, t = k % 25, ky = t / 5, kx = t % 5;
      int y = 2 * py + yv + ky - 2, xx = x0 + n + kx - 2;
      if (y >= 0 && y < HH && xx >= 0 && xx < WW)
        v = f2bf(x[((b * INC + ci) * HH + y) * WW + xx]);
    }
    bstage[i] = v;
  }
  __syncthreads();
  const bool hi = lane >= 16;
  const int n = lane & 15;
  const int cobase = wave * 16;
  const unsigned short* arow =
      w1r + (size_t)(l * OUTC + cobase + n) * 96 + (hi ? 8 : 0);
  const ushort8* bl = reinterpret_cast<const ushort8*>(bstage);
  v8f c0 = {}, c1 = {};
#pragma unroll
  for (int kk = 0; kk < 3; ++kk) {
    v16bf afr, bf0, bf1;
    const ushort8* ap = reinterpret_cast<const ushort8*>(arow + kk * 32);
    fill_afrag(afr, ap[0], ap[2]);
    int i0 = (kk * 32 + lane) * 2;                    // ushort8 units
    int i1 = ((3 + kk) * 32 + lane) * 2;
    fill_afrag(bf0, bl[i0], bl[i0 + 1]);
    fill_afrag(bf1, bl[i1], bl[i1 + 1]);
    c0 = __builtin_amdgcn_wmma_f32_16x16x32_bf16(false, afr, false, bf0,
                                                 (short)0, c0, false, false);
    c1 = __builtin_amdgcn_wmma_f32_16x16x32_bf16(false, afr, false, bf1,
                                                 (short)0, c1, false, false);
  }
#pragma unroll
  for (int v = 0; v < 8; ++v) {
    float f = fmaxf(fmaxf(c0[v], c1[v]), 0.0f);       // vertical pool + relu
    f = fmaxf(f, __shfl_xor(f, 1, 32));               // horizontal pool
    if ((lane & 1) == 0) {
      int co = cobase + v + (hi ? 8 : 0);
      int px = (x0 >> 1) + (n >> 1);
      h[(((size_t)(l * NB + b) * OUTC + co) * H2 + py) * W2 + px] = f2bf(f);
    }
  }
}

// ===========================================================================
// conv2 + relu + global spatial max (dominant: 107 GFLOP).
// K reordered tap-major (t=ky*5+kx, then ci): LDS patch stored channel-fastest
// [row(5)][col(36)][ci(64)] so every B fragment is 2 contiguous ds_load_b128;
// A rows (w2r) are K-contiguous -> 2 global b128.  Block covers a full 32-wide
// row (2 N-tiles share one A fragment + one LDS stage): 100 wmma / wave.
// ===========================================================================
__global__ __launch_bounds__(128)
void conv2_kernel(const unsigned short* __restrict__ h,
                  const unsigned short* __restrict__ w2r,
                  float* __restrict__ feat) {
  const int l = blockIdx.z, b = blockIdx.y;
  const int yo = blockIdx.x;                          // output row 0..31
  const int tid = threadIdx.x;
  const int lane = tid & 31, wave = tid >> 5;
  __shared__ unsigned short patch[5 * 36 * 64];       // 23,040 B
  const size_t hbase = (size_t)(l * NB + b) * OUTC * (H2 * W2);
  for (int i = tid; i < 5 * 36 * 64; i += 128) {
    int ci = i & 63, cr = i >> 6;                     // cr = row*36 + col
    int col = cr % 36, row = cr / 36;
    int y = yo - 2 + row, xx = col - 2;
    unsigned short v = 0;
    if (y >= 0 && y < H2 && xx >= 0 && xx < W2)
      v = h[hbase + (size_t)ci * (H2 * W2) + y * W2 + xx];
    patch[i] = v;
  }
  __syncthreads();
  const bool hi = lane >= 16;
  const int n = lane & 15;
  const int co = wave * 16 + n;
  const unsigned short* arow =
      w2r + (size_t)(l * OUTC + co) * 1600 + (hi ? 16 : 0);
  const ushort8* pl = reinterpret_cast<const ushort8*>(patch);
  v8f acc0 = {}, acc1 = {};
  const int hioff = hi ? 2 : 0;                       // +16 elems in ushort8 units
  for (int kk = 0; kk < 50; ++kk) {
    const int t = kk >> 1;                            // tap (uniform -> SALU)
    const int cb8 = (kk & 1) * 4 + hioff;             // channel base, ushort8 units
    const int ky = t / 5, kx = t % 5;
    v16bf afr, bf0, bf1;
    const ushort8* ap = reinterpret_cast<const ushort8*>(arow + kk * 32);
    fill_afrag(afr, ap[0], ap[1]);
    int base0 = (ky * 36 + n + kx) * 8 + cb8;         // tile0: x = n - 2 + kx
    int base1 = base0 + 16 * 8;                       // tile1: +16 columns
    fill_afrag(bf0, pl[base0], pl[base0 + 1]);
    fill_afrag(bf1, pl[base1], pl[base1 + 1]);
    acc0 = __builtin_amdgcn_wmma_f32_16x16x32_bf16(false, afr, false, bf0,
                                                   (short)0, acc0, false, false);
    acc1 = __builtin_amdgcn_wmma_f32_16x16x32_bf16(false, afr, false, bf1,
                                                   (short)0, acc1, false, false);
  }
#pragma unroll
  for (int v = 0; v < 8; ++v) {
    float f = fmaxf(fmaxf(acc0[v], acc1[v]), 0.0f);   // relu + max over 2 tiles
    f = fmaxf(f, __shfl_xor(f, 1, 32));               // max over 16 N positions
    f = fmaxf(f, __shfl_xor(f, 2, 32));
    f = fmaxf(f, __shfl_xor(f, 4, 32));
    f = fmaxf(f, __shfl_xor(f, 8, 32));
    if (n == 0) {                                     // relu => f>=0, int-max OK
      int c2 = wave * 16 + v + (hi ? 8 : 0);
      atomicMax((int*)&feat[(l * NB + b) * OUTC + c2], __float_as_int(f));
    }
  }
}

// ===========================================================================
// Per-batch MLP over all 16 leaves + mixture combine (deterministic, tiny).
// ===========================================================================
__global__ __launch_bounds__(128)
void mlp_kernel(const float* __restrict__ feat, const float* __restrict__ w1,
                const float* __restrict__ b1, const float* __restrict__ w2,
                const float* __restrict__ b2, const float* __restrict__ mix,
                float* __restrict__ out) {
  const int b = blockIdx.x, t = threadIdx.x;
  __shared__ float hid[LEAFW];
  __shared__ float fsh[OUTC];
  float oacc = 0.0f;
  for (int l = 0; l < NLEAF; ++l) {
    if (t < OUTC) fsh[t] = feat[(l * NB + b) * OUTC + t];
    __syncthreads();
    float s = b1[l * LEAFW + t];
    for (int c = 0; c < OUTC; ++c) s += fsh[c] * w1[(l * OUTC + c) * LEAFW + t];
    hid[t] = s;
    __syncthreads();
    if (t < OUTW) {
      float a = b2[l * OUTW + t];
      for (int j = 0; j < LEAFW; ++j) a += hid[j] * w2[(l * LEAFW + j) * OUTW + t];
      oacc += mix[b * NLEAF + l] * a;
    }
    __syncthreads();
  }
  if (t < OUTW) out[b * OUTW + t] = oacc;
}

// ===========================================================================
extern "C" void kernel_launch(void* const* d_in, const int* in_sizes, int n_in,
                              void* d_out, int out_size, void* d_ws, size_t ws_size,
                              hipStream_t stream) {
  const float* x   = (const float*)d_in[0];
  const float* nw  = (const float*)d_in[1];
  const float* nb  = (const float*)d_in[2];
  const float* cw1 = (const float*)d_in[3];
  const float* cw2 = (const float*)d_in[4];
  const float* w1  = (const float*)d_in[5];
  const float* b1  = (const float*)d_in[6];
  const float* w2  = (const float*)d_in[7];
  const float* b2  = (const float*)d_in[8];
  float* out = (float*)d_out;

  char* ws = (char*)d_ws;
  unsigned short* w1r = (unsigned short*)(ws + OFF_W1R);
  unsigned short* w2r = (unsigned short*)(ws + OFF_W2R);
  unsigned short* h   = (unsigned short*)(ws + OFF_H);
  float* feat = (float*)(ws + OFF_FEAT);
  float* mix  = (float*)(ws + OFF_MIX);

  prep_kernel<<<2048, 256, 0, stream>>>(cw1, cw2, w1r, w2r, feat);
  route_kernel<<<NB, 256, 0, stream>>>(x, nw, nb, mix);
  conv1_kernel<<<dim3(32 * 4, NB, NLEAF), 128, 0, stream>>>(x, w1r, h);
  conv2_kernel<<<dim3(32, NB, NLEAF), 128, 0, stream>>>(h, w2r, feat);
  mlp_kernel<<<NB, 128, 0, stream>>>(feat, w1, b1, w2, b2, mix, out);
}
